// MemoryEfficientAttention_19335942766927
// MI455X (gfx1250) — compile-verified, matrix operand order checked by reading, not run on hardware
//
#include <hip/hip_runtime.h>

#define BATCH  2
#define NSEQ   2048
#define CDIM   1024
#define NHEADS 16
#define HDIM   64
#define BN     (BATCH * NSEQ)   // 4096 total rows

typedef __attribute__((ext_vector_type(16))) _Float16 v16h;
typedef __attribute__((ext_vector_type(8)))  _Float16 v8h;
typedef __attribute__((ext_vector_type(8)))  float    v8f;

// 0.125 (=1/sqrt(HDIM)) * log2(e): folded into the Q projection so the
// attention kernel can use v_exp_f32 (2^x) directly with no per-element mul.
#define QSCALE 0.1803368801f

// ---------------------------------------------------------------------------
// Fragment builder matching the CDNA5 16-bit A/B lane layout:
//   lanes 0-15 : K elements {k0+0..7, k0+16..23}
//   lanes 16-31: K elements {k0+8..15, k0+24..31}
// Caller passes a pointer already offset by (hi ? 8 : 0).
// ---------------------------------------------------------------------------
__device__ __forceinline__ v16h frag_from_h16(const _Float16* p) {
    v8h lo = *(const v8h*)p;
    v8h hh = *(const v8h*)(p + 16);
    v16h r;
#pragma unroll
    for (int i = 0; i < 8; ++i) { r[i] = lo[i]; r[8 + i] = hh[i]; }
    return r;
}

// ---------------------------------------------------------------------------
// f32 -> f16 bulk convert (8 elements / thread, b128 in / b128 out)
// ---------------------------------------------------------------------------
__global__ __launch_bounds__(256) void cvt_kernel(const float* __restrict__ in,
                                                  _Float16* __restrict__ out,
                                                  int n8) {
    const int i = blockIdx.x * 256 + threadIdx.x;
    if (i >= n8) return;
    const float4 a = ((const float4*)in)[2 * i + 0];
    const float4 b = ((const float4*)in)[2 * i + 1];
    v8h r;
    r[0] = (_Float16)a.x; r[1] = (_Float16)a.y;
    r[2] = (_Float16)a.z; r[3] = (_Float16)a.w;
    r[4] = (_Float16)b.x; r[5] = (_Float16)b.y;
    r[6] = (_Float16)b.z; r[7] = (_Float16)b.w;
    ((v8h*)out)[i] = r;
}

// ---------------------------------------------------------------------------
// Transposed projection GEMM: computes Out^T = W @ X^T  (i.e. Out = X @ W^T),
// + bias, * oscale.  One wave: 128 output channels (j) x 16 rows (m).
// The D-matrix C-layout then has j running along registers -> contiguous
// epilogue stores; bias is indexed along registers.
// MODE 0: f16 out, head-major   [B,H,N,D]   (Q with oscale=QSCALE, K)
// MODE 1: f16 out, d-transposed [B,H,D,N]   (V)
// MODE 2: f32 out, plain        [BN,CDIM]   (final output projection)
// ---------------------------------------------------------------------------
template <int MODE>
__global__ __launch_bounds__(32) void proj_kernel(const _Float16* __restrict__ X,
                                                  const _Float16* __restrict__ W,
                                                  const float* __restrict__ bias,
                                                  void* __restrict__ outv,
                                                  float oscale) {
    const int lane = threadIdx.x;
    const int col  = lane & 15;
    const int hi   = lane >> 4;
    const int m0   = blockIdx.x * 16;    // row tile over BN (lanes)
    const int n0   = blockIdx.y * 128;   // output-channel tile (registers)

    v8f c[8] = {v8f{}, v8f{}, v8f{}, v8f{}, v8f{}, v8f{}, v8f{}, v8f{}};

    for (int k0 = 0; k0 < CDIM; k0 += 32) {
        // B = X^T fragment: lane holds column m, contiguous in k
        v16h bX = frag_from_h16(X + (size_t)(m0 + col) * CDIM + k0 + hi * 8);
#pragma unroll
        for (int g = 0; g < 8; ++g) {
            // A = W fragment: lane holds row j, contiguous in k
            v16h aW = frag_from_h16(W + (size_t)(n0 + g * 16 + col) * CDIM + k0 + hi * 8);
            c[g] = __builtin_amdgcn_wmma_f32_16x16x32_f16(
                false, aW, false, bX, (short)0, c[g], false, false);
        }
    }

    // Epilogue: Out^T C-layout -> lane holds fixed m; register i is channel
    // j0+i with j0 = n0 + g*16 + hi*8 (8 contiguous channels per store).
    const int m = m0 + col;
    const int b = m >> 11;               // m / NSEQ
    const int n = m & 2047;              // m % NSEQ
#pragma unroll
    for (int g = 0; g < 8; ++g) {
        const int j0 = n0 + g * 16 + hi * 8;
        float vals[8];
#pragma unroll
        for (int i = 0; i < 8; ++i) vals[i] = (c[g][i] + bias[j0 + i]) * oscale;
        if (MODE == 0) {
            const int h = j0 >> 6, d0 = j0 & 63;
            v8h r;
#pragma unroll
            for (int i = 0; i < 8; ++i) r[i] = (_Float16)vals[i];
            *(v8h*)((_Float16*)outv +
                    (((size_t)b * NHEADS + h) * NSEQ + n) * HDIM + d0) = r;
        } else if (MODE == 1) {
            const int h = j0 >> 6, d0 = j0 & 63;
#pragma unroll
            for (int i = 0; i < 8; ++i)
                ((_Float16*)outv)[(((size_t)b * NHEADS + h) * HDIM + d0 + i) * NSEQ + n] =
                    (_Float16)vals[i];
        } else {
            float* op = (float*)outv + (size_t)m * CDIM + j0;
            *(float4*)(op + 0) = make_float4(vals[0], vals[1], vals[2], vals[3]);
            *(float4*)(op + 4) = make_float4(vals[4], vals[5], vals[6], vals[7]);
        }
    }
}

// ---------------------------------------------------------------------------
// Attention, transposed formulation (no LDS, no barriers):
//   S^T = K · Q^T      (A = K tile, B = Q^T tile; both contiguous loads)
//   P^T = exp2(S^T)    (C-layout of S^T IS the B-layout of P^T for step 2)
//   O^T += V^T · P^T   (A = V^T from the pre-transposed Vt tensor)
// One wave: 32 q rows (two 16-col subtiles of S^T), kv chunks of 32.
// Row sums of P are per-lane scalars (lane = fixed q); one xor-16 shuffle at
// the end.  O^T epilogue stores are contiguous v8h.
// ---------------------------------------------------------------------------
__global__ __launch_bounds__(32) void attn_kernel(const _Float16* __restrict__ Qh,
                                                  const _Float16* __restrict__ Kh,
                                                  const _Float16* __restrict__ Vt,
                                                  _Float16* __restrict__ AO) {
    const int lane = threadIdx.x;
    const int col  = lane & 15;
    const int hi   = lane >> 4;
    const int q0   = blockIdx.x * 32;
    const int bh   = blockIdx.y;              // b*NHEADS + h
    const int b    = bh >> 4;
    const int h    = bh & 15;

    // B = Q^T fragments: [q-subtile][d-fragment]; lane holds q column
    v16h bQ[2][2];
#pragma unroll
    for (int qt = 0; qt < 2; ++qt)
#pragma unroll
        for (int f = 0; f < 2; ++f)
            bQ[qt][f] = frag_from_h16(
                Qh + ((size_t)bh * NSEQ + q0 + qt * 16 + col) * HDIM + f * 32 + hi * 8);

    float lrow[2] = {0.0f, 0.0f};
    v8f o[2][4];   // O^T accumulators: [q-subtile][d-group]
#pragma unroll
    for (int qt = 0; qt < 2; ++qt)
#pragma unroll
        for (int g = 0; g < 4; ++g) o[qt][g] = v8f{};

    for (int chunk = 0; chunk < NSEQ; chunk += 32) {
        // ---- S^T tiles: s[t][qt] = (K rows chunk+t*16..+15) x (q subtile) ----
        v8f s[2][2];
#pragma unroll
        for (int t = 0; t < 2; ++t) {
            const _Float16* kb =
                Kh + ((size_t)bh * NSEQ + chunk + t * 16 + col) * HDIM + hi * 8;
            v16h aK0 = frag_from_h16(kb);
            v16h aK1 = frag_from_h16(kb + 32);
#pragma unroll
            for (int qt = 0; qt < 2; ++qt) {
                v8f acc = {};
                acc = __builtin_amdgcn_wmma_f32_16x16x32_f16(
                    false, aK0, false, bQ[qt][0], (short)0, acc, false, false);
                acc = __builtin_amdgcn_wmma_f32_16x16x32_f16(
                    false, aK1, false, bQ[qt][1], (short)0, acc, false, false);
                s[t][qt] = acc;
            }
        }

        // ---- P^T = 2^(S^T): pack straight into B-layout fragments ----
        // B fragment elements {0..7} = kv tile t=0 regs, {8..15} = t=1 regs.
        v16h pb[2];
#pragma unroll
        for (int qt = 0; qt < 2; ++qt) {
            float acc = 0.0f;
#pragma unroll
            for (int i = 0; i < 8; ++i) {
                const float p0 = __builtin_amdgcn_exp2f(s[0][qt][i]);
                const float p1 = __builtin_amdgcn_exp2f(s[1][qt][i]);
                acc += p0 + p1;
                pb[qt][i]     = (_Float16)p0;
                pb[qt][8 + i] = (_Float16)p1;
            }
            lrow[qt] += acc;
        }

        // ---- O^T += V^T x P^T  (Vt: kv contiguous -> A fragments) ----
#pragma unroll
        for (int g = 0; g < 4; ++g) {
            v16h aV = frag_from_h16(
                Vt + ((size_t)bh * HDIM + g * 16 + col) * NSEQ + chunk + hi * 8);
#pragma unroll
            for (int qt = 0; qt < 2; ++qt)
                o[qt][g] = __builtin_amdgcn_wmma_f32_16x16x32_f16(
                    false, aV, false, pb[qt], (short)0, o[qt][g], false, false);
        }
    }

    // ---- finalize: combine the two kv half-sums, normalize, store O ----
#pragma unroll
    for (int qt = 0; qt < 2; ++qt) {
        const float l   = lrow[qt] + __shfl_xor(lrow[qt], 16, 32);
        const float inv = 1.0f / l;
        const int   qrow = q0 + qt * 16 + col;           // lane = q column
        _Float16* rowp = AO + ((size_t)b * NSEQ + qrow) * CDIM + h * HDIM;
#pragma unroll
        for (int g = 0; g < 4; ++g) {
            v8h r;
#pragma unroll
            for (int i = 0; i < 8; ++i) r[i] = (_Float16)(o[qt][g][i] * inv);
            *(v8h*)(rowp + g * 16 + hi * 8) = r;          // 8 contiguous d
        }
    }
}

// ---------------------------------------------------------------------------
extern "C" void kernel_launch(void* const* d_in, const int* in_sizes, int n_in,
                              void* d_out, int out_size, void* d_ws, size_t ws_size,
                              hipStream_t stream) {
    const float* q  = (const float*)d_in[0];
    const float* k  = (const float*)d_in[1];
    const float* v  = (const float*)d_in[2];
    const float* Wq = (const float*)d_in[3];
    const float* bq = (const float*)d_in[4];
    const float* Wk = (const float*)d_in[5];
    const float* bk = (const float*)d_in[6];
    const float* Wv = (const float*)d_in[7];
    const float* bv = (const float*)d_in[8];
    const float* Wo = (const float*)d_in[9];
    const float* bo = (const float*)d_in[10];
    float* out = (float*)d_out;

    // Workspace layout (f16 tensors)
    char* ws = (char*)d_ws;
    const size_t xbytes = (size_t)BN * CDIM * sizeof(_Float16);     // 8 MB
    const size_t wbytes = (size_t)CDIM * CDIM * sizeof(_Float16);   // 2 MB
    _Float16* Xq  = (_Float16*)(ws);
    _Float16* Xk  = (_Float16*)(ws + 1 * xbytes);
    _Float16* Xv  = (_Float16*)(ws + 2 * xbytes);
    _Float16* Wqh = (_Float16*)(ws + 3 * xbytes);
    _Float16* Wkh = (_Float16*)(ws + 3 * xbytes + 1 * wbytes);
    _Float16* Wvh = (_Float16*)(ws + 3 * xbytes + 2 * wbytes);
    _Float16* Woh = (_Float16*)(ws + 3 * xbytes + 3 * wbytes);
    char* base2 = ws + 3 * xbytes + 4 * wbytes;
    _Float16* Qh = (_Float16*)(base2);
    _Float16* Kh = (_Float16*)(base2 + 1 * xbytes);
    _Float16* Vt = (_Float16*)(base2 + 2 * xbytes);
    _Float16* AO = (_Float16*)(base2 + 3 * xbytes);

    // 1) bulk f32 -> f16 conversion of activations and weights
    const int xn8 = (BN * CDIM) / 8;       // 524288
    const int wn8 = (CDIM * CDIM) / 8;     // 131072
    cvt_kernel<<<(xn8 + 255) / 256, 256, 0, stream>>>(q,  Xq,  xn8);
    cvt_kernel<<<(xn8 + 255) / 256, 256, 0, stream>>>(k,  Xk,  xn8);
    cvt_kernel<<<(xn8 + 255) / 256, 256, 0, stream>>>(v,  Xv,  xn8);
    cvt_kernel<<<(wn8 + 255) / 256, 256, 0, stream>>>(Wq, Wqh, wn8);
    cvt_kernel<<<(wn8 + 255) / 256, 256, 0, stream>>>(Wk, Wkh, wn8);
    cvt_kernel<<<(wn8 + 255) / 256, 256, 0, stream>>>(Wv, Wvh, wn8);
    cvt_kernel<<<(wn8 + 255) / 256, 256, 0, stream>>>(Wo, Woh, wn8);

    // 2) projections (1/sqrt(HDIM) * log2(e) folded into Q)
    const dim3 gProj(BN / 16, CDIM / 128);
    proj_kernel<0><<<gProj, 32, 0, stream>>>(Xq, Wqh, bq, (void*)Qh, QSCALE);
    proj_kernel<0><<<gProj, 32, 0, stream>>>(Xk, Wkh, bk, (void*)Kh, 1.0f);
    proj_kernel<1><<<gProj, 32, 0, stream>>>(Xv, Wvh, bv, (void*)Vt, 1.0f);

    // 3) attention
    attn_kernel<<<dim3(NSEQ / 32, BATCH * NHEADS), 32, 0, stream>>>(Qh, Kh, Vt, AO);

    // 4) output projection -> f32 d_out
    proj_kernel<2><<<gProj, 32, 0, stream>>>(AO, Woh, bo, (void*)out, 1.0f);
}
